// QSSM_35605278884497
// MI455X (gfx1250) — compile-verified, matrix operand order checked by reading, not run on hardware
//
#include <hip/hip_runtime.h>
#include <math.h>

// ---------------------------------------------------------------------------
// Types for CDNA5 WMMA (gfx1250, wave32)
// ---------------------------------------------------------------------------
typedef __attribute__((ext_vector_type(16))) __bf16 bf16x16;
typedef __attribute__((ext_vector_type(8)))  __bf16 bf16x8;
typedef __attribute__((ext_vector_type(8)))  float  f32x8;

#define DEV __device__ __forceinline__

// Load a 16x32 bf16 A-fragment (or a B-fragment from an (N,K) row-major
// weight, which has the identical per-lane pattern with N in place of M).
// Per ISA 7.12.2: lane l -> row (l&15); K indices = kb..kb+7 and kb+16..kb+23
// where kb = (l>=16 ? 8 : 0). Both runs are 16-byte contiguous loads.
DEV bf16x16 load_frag_bf16(const __bf16* __restrict__ base, int ld,
                           int row0, int k0, int lane) {
  int m  = lane & 15;
  int kb = (lane >> 4) << 3;
  const __bf16* p = base + (size_t)(row0 + m) * (size_t)ld + (size_t)(k0 + kb);
  bf16x8 lo = *(const bf16x8*)p;
  bf16x8 hi = *(const bf16x8*)(p + 16);
  bf16x16 r;
#pragma unroll
  for (int e = 0; e < 8; ++e) { r[e] = lo[e]; r[e + 8] = hi[e]; }
  return r;
}

DEV f32x8 wmma_bf16(bf16x16 a, bf16x16 b, f32x8 c) {
  return __builtin_amdgcn_wmma_f32_16x16x32_bf16(
      /*neg_a=*/false, a, /*neg_b=*/false, b,
      /*c_mod=*/(short)0, c, /*reuse_a=*/false, /*reuse_b=*/false);
}

// ---------------------------------------------------------------------------
// Problem constants
// ---------------------------------------------------------------------------
#define BB      4
#define LL      4096
#define DM      256
#define DI      512
#define DS      16
#define DTR     16
#define NQ      256
#define HEADS   4
#define HD      64
#define ROWS    (BB * LL)        // 16384
#define LKV     (2 * LL)         // 8192
#define KVROWS  (BB * LKV)       // 32768
#define QROWS   (BB * NQ)        // 1024

// ---------------------------------------------------------------------------
// fp32 -> bf16 cast / zero fill
// ---------------------------------------------------------------------------
__global__ void __launch_bounds__(256)
qssm_cast_bf16(const float* __restrict__ s, __bf16* __restrict__ d, int n) {
  int i = blockIdx.x * blockDim.x + threadIdx.x;
  if (i < n) d[i] = (__bf16)s[i];
}

__global__ void __launch_bounds__(256)
qssm_zero(float* __restrict__ p, int n) {
  int i = blockIdx.x * blockDim.x + threadIdx.x;
  if (i < n) p[i] = 0.f;
}

// ---------------------------------------------------------------------------
// (optional residual-add +) LayerNorm over D=256. ONE WAVE PER BLOCK:
// row = blockIdx.x (scalar). Output row remap:
//   orow = (row / Lin) * ostride + off + (row % Lin)
// ---------------------------------------------------------------------------
__global__ void __launch_bounds__(32)
qssm_add_ln(const float* __restrict__ a, const float* __restrict__ res,
            const float* __restrict__ w, const float* __restrict__ bias,
            float* __restrict__ outF, __bf16* __restrict__ outB,
            int Lin, int ostride, int off) {
  int row = blockIdx.x;
  int lane = threadIdx.x;
  const float* ar = a + (size_t)row * DM;
  float v[8];
  float s = 0.f;
#pragma unroll
  for (int i = 0; i < 8; ++i) {
    int c = lane * 8 + i;
    float t = ar[c];
    if (res) t += res[(size_t)row * DM + c];
    v[i] = t; s += t;
  }
#pragma unroll
  for (int m = 1; m < 32; m <<= 1) s += __shfl_xor(s, m, 32);
  float mean = s * (1.f / DM);
  float q = 0.f;
#pragma unroll
  for (int i = 0; i < 8; ++i) { float dd = v[i] - mean; q += dd * dd; }
#pragma unroll
  for (int m = 1; m < 32; m <<= 1) q += __shfl_xor(q, m, 32);
  float inv = rsqrtf(q * (1.f / DM) + 1e-5f);
  int orow = (row / Lin) * ostride + off + (row % Lin);
#pragma unroll
  for (int i = 0; i < 8; ++i) {
    int c = lane * 8 + i;
    float o = (v[i] - mean) * inv * w[c] + bias[c];
    if (outF) outF[(size_t)orow * DM + c] = o;
    if (outB) outB[(size_t)orow * DM + c] = (__bf16)o;
  }
}

// ---------------------------------------------------------------------------
// WMMA GEMM:  C[M,N] = A[M,K] (bf16 row-major) x W[N,K]^T (bf16 row-major)
// ONE WAVE PER BLOCK computing a 16 x (NT*16) strip; grid sized exactly, so
// there are no guards and no divergent branches anywhere near the WMMAs.
// NT=8 for N % 128 == 0 (8 wmmas amortize each A-fragment load), NT=3 for the
// N=48 x-projection. MODE: 0 = f32 out, 1 = bf16 out, 2 = bf16 V^T scatter.
// ---------------------------------------------------------------------------
template <int NT, int MODE>
__global__ void __launch_bounds__(32)
qssm_gemm(const __bf16* __restrict__ A, const __bf16* __restrict__ W,
          float* __restrict__ outF, __bf16* __restrict__ outB,
          int M, int N, int K) {
  int mtiles = M >> 4;
  int mt = blockIdx.x % mtiles;          // scalar
  int ng = blockIdx.x / mtiles;          // scalar
  int lane = threadIdx.x;
  int row0 = mt << 4;
  int ncol0 = ng * (NT * 16);

  f32x8 acc[NT];
#pragma unroll
  for (int t = 0; t < NT; ++t)
#pragma unroll
    for (int r = 0; r < 8; ++r) acc[t][r] = 0.f;

  for (int k0 = 0; k0 < K; k0 += 32) {
    bf16x16 a = load_frag_bf16(A, K, row0, k0, lane);
#pragma unroll
    for (int t = 0; t < NT; ++t) {
      bf16x16 b = load_frag_bf16(W, K, ncol0 + (t << 4), k0, lane);
      acc[t] = wmma_bf16(a, b, acc[t]);
    }
  }

  int nlo = lane & 15;
  int mhi = (lane >> 4) << 3;   // C/D layout: M = r + 8*(lane>=16), N = lane&15
#pragma unroll
  for (int t = 0; t < NT; ++t) {
    int col = ncol0 + (t << 4) + nlo;
#pragma unroll
    for (int r = 0; r < 8; ++r) {
      int row = row0 + r + mhi;
      float v = acc[t][r];
      if constexpr (MODE == 0) {
        outF[(size_t)row * N + col] = v;
      } else if constexpr (MODE == 1) {
        outB[(size_t)row * N + col] = (__bf16)v;
      } else {
        int bb = row >> 13;            // row = b*LKV + lkv
        int lkv = row & (LKV - 1);
        int hh = col >> 6, e = col & 63;
        outB[(((size_t)(bb * HEADS + hh) * HD + e) << 13) + lkv] = (__bf16)v;
      }
    }
  }
}

// ---------------------------------------------------------------------------
// Causal depthwise conv (D_CONV=4) + bias + SiLU on the xi half of xz.
// ---------------------------------------------------------------------------
__global__ void __launch_bounds__(256)
qssm_conv_silu(const float* __restrict__ xz, const float* __restrict__ cw,
               const float* __restrict__ cb,
               float* __restrict__ xcf, __bf16* __restrict__ xcb) {
  int idx = blockIdx.x * blockDim.x + threadIdx.x;
  if (idx >= ROWS * DI) return;
  int row = idx >> 9;          // / DI
  int d = idx & (DI - 1);
  int l = row & (LL - 1);
  float acc = cb[d];
#pragma unroll
  for (int j = 0; j < 4; ++j) {
    int lj = l - 3 + j;
    if (lj >= 0) acc += xz[(size_t)(row - 3 + j) * (2 * DI) + d] * cw[d * 4 + j];
  }
  float y = acc / (1.f + __expf(-acc));  // SiLU
  xcf[idx] = y;
  xcb[idx] = (__bf16)y;
}

// ---------------------------------------------------------------------------
// Fused selective scan (ONE WAVE PER BLOCK): dt-projection (K=16) + softplus +
// state recurrence + C-contraction + D-skip + SiLU(z) gate. Lanes 0..15 hold
// one channel's 16 states, lanes 16..31 a second channel; width-16 shfl_xor
// does the state reductions. This 4096-step recurrence is the serial critical
// path of the whole problem, so each step's operands are prefetched ~48 steps
// ahead (global_prefetch_b8) to keep the dependency chain fed from L2.
// ---------------------------------------------------------------------------
__global__ void __launch_bounds__(32)
qssm_scan(const float* __restrict__ xc, const float* __restrict__ xdbl,
          const float* __restrict__ xz,
          const float* __restrict__ dt_w, const float* __restrict__ dt_b,
          const float* __restrict__ A_log, const float* __restrict__ Dp,
          __bf16* __restrict__ y_bf) {
  int lane = threadIdx.x;
  int s = lane & 15;
  int ch = blockIdx.x * 2 + (lane >> 4);   // 0..2047
  int b = ch >> 9;                         // / DI
  int d = ch & (DI - 1);
  float As  = -__expf(A_log[d * DS + s]);
  float dtw = dt_w[d * DTR + s];
  float dtb = dt_b[d];
  float Dd  = Dp[d];
  float h = 0.f;
  const int PFD = 48;   // prefetch distance (rows)
  for (int t = 0; t < LL; ++t) {
    int row = b * LL + t;
    // Prefetch the serial chain's future operands (harmless if past the end:
    // the addresses stay inside the workspace allocation).
    size_t prow = (size_t)(row + PFD);
    __builtin_prefetch(xdbl + prow * (DTR + 2 * DS), 0, 0);
    __builtin_prefetch(xc + prow * DI + d, 0, 0);
    __builtin_prefetch(xz + prow * (2 * DI) + DI + d, 0, 0);

    const float* xr = xdbl + (size_t)row * (DTR + 2 * DS);
    // dt = softplus(x_dbl[:,:16] . dt_w[d] + dt_b[d])
    float sp = xr[s] * dtw;
    sp += __shfl_xor(sp, 1, 16); sp += __shfl_xor(sp, 2, 16);
    sp += __shfl_xor(sp, 4, 16); sp += __shfl_xor(sp, 8, 16);
    float v = sp + dtb;
    float dt = (v > 20.f) ? v : log1pf(__expf(v));
    float x_t = xc[(size_t)row * DI + d];
    float Bv = xr[DTR + s];
    float Cv = xr[DTR + DS + s];
    h = h * __expf(dt * As) + dt * x_t * Bv;
    float yc = h * Cv;
    yc += __shfl_xor(yc, 1, 16); yc += __shfl_xor(yc, 2, 16);
    yc += __shfl_xor(yc, 4, 16); yc += __shfl_xor(yc, 8, 16);
    if (s == 0) {
      float z = xz[(size_t)row * (2 * DI) + DI + d];
      float sz = z / (1.f + __expf(-z));
      y_bf[(size_t)row * DI + d] = (__bf16)((yc + x_t * Dd) * sz);
    }
  }
}

// ---------------------------------------------------------------------------
// Flash attention, ONE WAVE PER BLOCK: block = (b, h, 16-query tile), all
// scalar. 32-key chunks: S = Q(16x64) K^T via 4 wmmas; online softmax in
// C-layout registers (row M = r + 8*(lane>=16)); P goes through a wave-
// private LDS tile to be re-laid out as an A-fragment; O += P(16x32) V(32x64)
// via 4 wmmas against pre-transposed V.
// ---------------------------------------------------------------------------
__global__ void __launch_bounds__(32)
qssm_flash_attn(const __bf16* __restrict__ Q, const __bf16* __restrict__ Kt,
                const __bf16* __restrict__ Vt, float* __restrict__ O) {
  __shared__ float plds[16][32];
  int lane = threadIdx.x;
  int wave = blockIdx.x;
  int qt = wave & 15;                 // scalar
  int h = (wave >> 4) & (HEADS - 1);  // scalar
  int b = wave >> 6;                  // scalar

  int m  = lane & 15;
  int g  = lane >> 4;
  int kb = g << 3;

  bf16x16 qa0 = load_frag_bf16(Q, DM, b * NQ + qt * 16, h * HD,      lane);
  bf16x16 qa1 = load_frag_bf16(Q, DM, b * NQ + qt * 16, h * HD + 32, lane);

  f32x8 o[4];
  float mrow[8], lrow[8];
#pragma unroll
  for (int t = 0; t < 4; ++t)
#pragma unroll
    for (int r = 0; r < 8; ++r) o[t][r] = 0.f;
#pragma unroll
  for (int r = 0; r < 8; ++r) { mrow[r] = -1e30f; lrow[r] = 0.f; }

  const float sc = 0.125f;     // 1/sqrt(64)

  for (int kc = 0; kc < LKV; kc += 32) {
    f32x8 s0, s1;
#pragma unroll
    for (int r = 0; r < 8; ++r) { s0[r] = 0.f; s1[r] = 0.f; }

    bf16x16 k0a = load_frag_bf16(Kt, DM, b * LKV + kc,      h * HD, lane);
    bf16x16 k1a = load_frag_bf16(Kt, DM, b * LKV + kc + 16, h * HD, lane);
    s0 = wmma_bf16(qa0, k0a, s0);
    s1 = wmma_bf16(qa0, k1a, s1);
    k0a = load_frag_bf16(Kt, DM, b * LKV + kc,      h * HD + 32, lane);
    k1a = load_frag_bf16(Kt, DM, b * LKV + kc + 16, h * HD + 32, lane);
    s0 = wmma_bf16(qa1, k0a, s0);
    s1 = wmma_bf16(qa1, k1a, s1);

#pragma unroll
    for (int r = 0; r < 8; ++r) {
      float a0 = s0[r] * sc;
      float a1 = s1[r] * sc;
      float mx = fmaxf(a0, a1);
      mx = fmaxf(mx, __shfl_xor(mx, 1, 16));
      mx = fmaxf(mx, __shfl_xor(mx, 2, 16));
      mx = fmaxf(mx, __shfl_xor(mx, 4, 16));
      mx = fmaxf(mx, __shfl_xor(mx, 8, 16));
      float mnew = fmaxf(mrow[r], mx);
      float alpha = __expf(mrow[r] - mnew);
      float p0 = __expf(a0 - mnew);
      float p1 = __expf(a1 - mnew);
      float rs = p0 + p1;
      rs += __shfl_xor(rs, 1, 16); rs += __shfl_xor(rs, 2, 16);
      rs += __shfl_xor(rs, 4, 16); rs += __shfl_xor(rs, 8, 16);
      lrow[r] = lrow[r] * alpha + rs;
      mrow[r] = mnew;
#pragma unroll
      for (int t = 0; t < 4; ++t) o[t][r] *= alpha;
      plds[r + 8 * g][m]      = p0;    // [query row][key]
      plds[r + 8 * g][m + 16] = p1;
    }
    __syncthreads();   // LDS fence (single-wave block; DS ops are in-order)

    bf16x16 pa;
#pragma unroll
    for (int e = 0; e < 8; ++e) {
      pa[e]     = (__bf16)plds[m][kb + e];
      pa[e + 8] = (__bf16)plds[m][kb + 16 + e];
    }
#pragma unroll
    for (int t = 0; t < 4; ++t) {
      bf16x16 vb = load_frag_bf16(Vt, LKV, (b * HEADS + h) * HD + t * 16, kc, lane);
      o[t] = wmma_bf16(pa, vb, o[t]);
    }
    __syncthreads();
  }

#pragma unroll
  for (int t = 0; t < 4; ++t)
#pragma unroll
    for (int r = 0; r < 8; ++r) {
      int qrow = b * NQ + qt * 16 + r + 8 * g;
      int col  = h * HD + t * 16 + m;
      O[(size_t)qrow * DM + col] = o[t][r] / lrow[r];
    }
}

// ---------------------------------------------------------------------------
// Feedback-pool: LN each concat row ([attn_q ; queries]) and accumulate the
// mean. ONE WAVE PER BLOCK.
// ---------------------------------------------------------------------------
__global__ void __launch_bounds__(32)
qssm_pool_ln(const float* __restrict__ attn_q, const float* __restrict__ queries,
             const float* __restrict__ w, const float* __restrict__ bias,
             float* __restrict__ pooled) {
  int row = blockIdx.x;               // 0 .. 2047
  int lane = threadIdx.x;
  int b = row >> 9, j = row & 511;
  const float* src = (j < NQ) ? attn_q  + ((size_t)b * NQ + j) * DM
                              : queries + ((size_t)b * NQ + (j - NQ)) * DM;
  float v[8];
  float s = 0.f;
#pragma unroll
  for (int i = 0; i < 8; ++i) { v[i] = src[lane * 8 + i]; s += v[i]; }
#pragma unroll
  for (int mk = 1; mk < 32; mk <<= 1) s += __shfl_xor(s, mk, 32);
  float mean = s * (1.f / DM);
  float q = 0.f;
#pragma unroll
  for (int i = 0; i < 8; ++i) { float dd = v[i] - mean; q += dd * dd; }
#pragma unroll
  for (int mk = 1; mk < 32; mk <<= 1) q += __shfl_xor(q, mk, 32);
  float inv = rsqrtf(q * (1.f / DM) + 1e-5f);
#pragma unroll
  for (int i = 0; i < 8; ++i) {
    int c = lane * 8 + i;
    float o = (v[i] - mean) * inv * w[c] + bias[c];
    atomicAdd(&pooled[b * DM + c], o * (1.f / (2.f * NQ)));
  }
}

__global__ void __launch_bounds__(256)
qssm_cw(const float* __restrict__ pooled, const float* __restrict__ pw,
        const float* __restrict__ pb, float* __restrict__ cw) {
  int i = blockIdx.x * blockDim.x + threadIdx.x;
  if (i >= BB * DM) return;
  int b = i >> 8, c = i & 255;
  float acc = pb[c];
  for (int k = 0; k < DM; ++k) acc += pooled[b * DM + k] * pw[c * DM + k];
  cw[i] = fmaxf(acc, 0.f);
}

__global__ void __launch_bounds__(256)
qssm_final(const float* __restrict__ attn_q, const float* __restrict__ cw,
           const float* __restrict__ queries, const float* __restrict__ gamma,
           float* __restrict__ out) {
  int i = blockIdx.x * blockDim.x + threadIdx.x;
  if (i >= QROWS * DM) return;
  int b = i >> 16;          // / (NQ*DM)
  int c = i & 255;
  out[i] = gamma[0] * attn_q[i] * cw[b * DM + c] + queries[i];
}

// ---------------------------------------------------------------------------
// Host: workspace layout (bytes). Mamba scratch is reused by attention stage;
// kv_bf persists at a high offset.
// ---------------------------------------------------------------------------
static const size_t WS_XZ    = 0;                                     // 64 MiB
static const size_t WS_XC    = WS_XZ    + (size_t)ROWS * 1024 * 4;    // 32 MiB
static const size_t WS_XCBF  = WS_XC    + (size_t)ROWS * DI * 4;      // 16 MiB
static const size_t WS_XDBL  = WS_XCBF  + (size_t)ROWS * DI * 2;      // 3 MiB
static const size_t WS_YBF   = WS_XDBL  + (size_t)ROWS * 48 * 4;      // 16 MiB
static const size_t WS_YPROJ = WS_YBF   + (size_t)ROWS * DI * 2;      // 16 MiB
static const size_t WS_HBF   = WS_YPROJ + (size_t)ROWS * DM * 4;      // 8 MiB
static const size_t WS_KV    = WS_HBF   + (size_t)ROWS * DM * 2;      // 16 MiB (persists)
static const size_t WS_WA    = WS_KV    + (size_t)KVROWS * DM * 2;    // mamba weights bf16

// Attention-stage overlays (inside the dead mamba scratch, < WS_KV)
static const size_t WS_QBF    = 0;
static const size_t WS_KBF    = (size_t)1  << 20;
static const size_t WS_VTBF   = (size_t)18 << 20;
static const size_t WS_ATTNF  = (size_t)35 << 20;
static const size_t WS_ATTNBF = (size_t)36 << 20;
static const size_t WS_APROJ  = (size_t)37 << 20;
static const size_t WS_ATTNQ  = (size_t)38 << 20;
static const size_t WS_QINBF  = (size_t)39 << 20;
static const size_t WS_POOL   = (size_t)40 << 20;
static const size_t WS_CWV    = ((size_t)40 << 20) + 8192;
static const size_t WS_WATTN  = (size_t)41 << 20;

// Host-side templated GEMM launcher: one wave (32-thread block) per 16x(NT*16)
// output strip; grid sized exactly -> no guards in the kernel.
template <int NT, int MODE>
static void launch_gemm(const __bf16* A, const __bf16* W, float* oF, __bf16* oB,
                        int M, int N, int K, hipStream_t stream) {
  int blocks = (M >> 4) * ((N + NT * 16 - 1) / (NT * 16));
  qssm_gemm<NT, MODE><<<blocks, 32, 0, stream>>>(A, W, oF, oB, M, N, K);
}

extern "C" void kernel_launch(void* const* d_in, const int* in_sizes, int n_in,
                              void* d_out, int out_size, void* d_ws, size_t ws_size,
                              hipStream_t stream) {
  (void)in_sizes; (void)n_in; (void)out_size; (void)ws_size;
  const float* x_i     = (const float*)d_in[0];
  const float* x_j     = (const float*)d_in[1];
  const float* queries = (const float*)d_in[2];
  const int PI = 3, PJ = 16, PA = 29, PF = 35;

  char* ws = (char*)d_ws;
  auto F  = [&](size_t off) { return (float*)(ws + off); };
  auto Bp = [&](size_t off) { return (__bf16*)(ws + off); };

  auto cast = [&](const float* s, __bf16* d, int n) {
    qssm_cast_bf16<<<(n + 255) / 256, 256, 0, stream>>>(s, d, n);
  };
  auto add_ln = [&](const float* a, const float* res, const float* w, const float* b,
                    float* oF, __bf16* oB, int rows, int Lin, int ostride, int off) {
    qssm_add_ln<<<rows, 32, 0, stream>>>(a, res, w, b, oF, oB, Lin, ostride, off);
  };

  __bf16* inw_bf = Bp(WS_WA);
  __bf16* xp_bf  = Bp(WS_WA + 524288);
  __bf16* ow_bf  = Bp(WS_WA + 524288 + 49152);
  __bf16* kv_bf  = Bp(WS_KV);

  auto run_mamba = [&](const float* x, void* const* prm, int kvoff) {
    const float* ln_w   = (const float*)prm[0];
    const float* ln_b   = (const float*)prm[1];
    const float* in_w   = (const float*)prm[2];
    const float* conv_w = (const float*)prm[3];
    const float* conv_b = (const float*)prm[4];
    const float* xproj  = (const float*)prm[5];
    const float* dt_w   = (const float*)prm[6];
    const float* dt_b   = (const float*)prm[7];
    const float* A_log  = (const float*)prm[8];
    const float* Dp     = (const float*)prm[9];
    const float* out_w  = (const float*)prm[10];
    const float* pln_w  = (const float*)prm[11];
    const float* pln_b  = (const float*)prm[12];

    cast(in_w, inw_bf, 2 * DI * DM);
    cast(xproj, xp_bf, 48 * DI);
    cast(out_w, ow_bf, DM * DI);

    add_ln(x, nullptr, ln_w, ln_b, nullptr, Bp(WS_HBF), ROWS, 1 << 30, 0, 0);
    launch_gemm<8, 0>(Bp(WS_HBF), inw_bf, F(WS_XZ), nullptr, ROWS, 2 * DI, DM, stream);
    qssm_conv_silu<<<(ROWS * DI) / 256, 256, 0, stream>>>(F(WS_XZ), conv_w, conv_b,
                                                          F(WS_XC), Bp(WS_XCBF));
    launch_gemm<3, 0>(Bp(WS_XCBF), xp_bf, F(WS_XDBL), nullptr, ROWS, 48, DI, stream);
    qssm_scan<<<1024, 32, 0, stream>>>(F(WS_XC), F(WS_XDBL), F(WS_XZ),
                                       dt_w, dt_b, A_log, Dp, Bp(WS_YBF));
    launch_gemm<8, 0>(Bp(WS_YBF), ow_bf, F(WS_YPROJ), nullptr, ROWS, DM, DI, stream);
    // f = LN(y_out + x); write bf16 into kv at rows b*8192 + kvoff + l
    add_ln(F(WS_YPROJ), x, pln_w, pln_b, nullptr, kv_bf, ROWS, LL, LKV, kvoff);
  };

  run_mamba(x_i, d_in + PI, 0);
  run_mamba(x_j, d_in + PJ, LL);

  // ---- attention stage ----
  const float* wq = (const float*)d_in[PA + 0];
  const float* wk = (const float*)d_in[PA + 1];
  const float* wv = (const float*)d_in[PA + 2];
  const float* wo = (const float*)d_in[PA + 3];
  const float* res_ln_w = (const float*)d_in[PA + 4];
  const float* res_ln_b = (const float*)d_in[PA + 5];
  const float* fb_ln_w  = (const float*)d_in[PF + 0];
  const float* fb_ln_b  = (const float*)d_in[PF + 1];
  const float* pool_w   = (const float*)d_in[PF + 2];
  const float* pool_b   = (const float*)d_in[PF + 3];
  const float* gamma    = (const float*)d_in[PF + 4];

  __bf16* wq_bf = Bp(WS_WATTN);
  __bf16* wk_bf = Bp(WS_WATTN + 131072);
  __bf16* wv_bf = Bp(WS_WATTN + 262144);
  __bf16* wo_bf = Bp(WS_WATTN + 393216);
  cast(wq, wq_bf, DM * DM);
  cast(wk, wk_bf, DM * DM);
  cast(wv, wv_bf, DM * DM);
  cast(wo, wo_bf, DM * DM);
  cast(queries, Bp(WS_QINBF), QROWS * DM);

  launch_gemm<8, 1>(Bp(WS_QINBF), wq_bf, nullptr, Bp(WS_QBF),  QROWS,  DM, DM, stream);
  launch_gemm<8, 1>(kv_bf,        wk_bf, nullptr, Bp(WS_KBF),  KVROWS, DM, DM, stream);
  launch_gemm<8, 2>(kv_bf,        wv_bf, nullptr, Bp(WS_VTBF), KVROWS, DM, DM, stream);  // V^T

  qssm_flash_attn<<<BB * HEADS * (NQ / 16), 32, 0, stream>>>(
      Bp(WS_QBF), Bp(WS_KBF), Bp(WS_VTBF), F(WS_ATTNF));

  cast(F(WS_ATTNF), Bp(WS_ATTNBF), QROWS * DM);
  launch_gemm<8, 0>(Bp(WS_ATTNBF), wo_bf, F(WS_APROJ), nullptr, QROWS, DM, DM, stream);
  add_ln(F(WS_APROJ), queries, res_ln_w, res_ln_b, F(WS_ATTNQ), nullptr,
         QROWS, 1 << 30, 0, 0);

  qssm_zero<<<4, 256, 0, stream>>>(F(WS_POOL), BB * DM);
  qssm_pool_ln<<<BB * 2 * NQ, 32, 0, stream>>>(F(WS_ATTNQ), queries,
                                               fb_ln_w, fb_ln_b, F(WS_POOL));
  qssm_cw<<<4, 256, 0, stream>>>(F(WS_POOL), pool_w, pool_b, F(WS_CWV));
  qssm_final<<<(QROWS * DM) / 256, 256, 0, stream>>>(F(WS_ATTNQ), F(WS_CWV),
                                                     queries, gamma, (float*)d_out);
}
